// MesaNet_79834852098143
// MI455X (gfx1250) — compile-verified, matrix operand order checked by reading, not run on hardware
//
#include <hip/hip_runtime.h>
#include <hip/hip_bf16.h>
#include <math.h>

// ---------------------------------------------------------------------------
// MesaNet forward for MI455X (gfx1250, wave32, WMMA + async LDS DMA).
//   B=2, T=1024, HID=1024, H=16, DK=DV=64, K=4, CG=15
// Pipeline:
//   0) cvt/transpose: hidden -> f16; Wq/Wk/Wv/Wo -> f16 transposed [N,K]
//   1) gemm_wmma: double-buffered GLOBAL_LOAD_ASYNC_TO_LDS_B128 staging,
//      4 waves x (64x64 tile) = 16 v_wmma_f32_16x16x32_f16 per wave per k-step
//   2) proj_ab / conv_silu_l2 elementwise kernels
//   3) scan_state: sequential decayed rank-1 prefix, snapshot every 32 steps
//   4) scan_cg: 1024 parallel chunk blocks: replay chunk, 15-iter CG, out+RMS (f16)
//   5) gemm_wmma: out = On16 @ Wo16T
// ---------------------------------------------------------------------------

#define Bv   2
#define Tv   1024
#define HIDv 1024
#define Hh   16
#define DK   64
#define CGIT 15
#define CHUNK 32
#define NCHUNK (Tv / CHUNK)

typedef __attribute__((ext_vector_type(16))) _Float16 v16h;
typedef __attribute__((ext_vector_type(8)))  float    v8f;

// Per-lane async DMA: 16 bytes global -> LDS, tracked by ASYNCcnt.
__device__ __forceinline__ void async_b128(unsigned lds_off, const _Float16* g) {
  asm volatile("global_load_async_to_lds_b128 %0, %1, off"
               :: "v"(lds_off), "v"((unsigned long long)(uintptr_t)g)
               : "memory");
}
template <int N>
__device__ __forceinline__ void wait_async_le() {
  asm volatile("s_wait_asynccnt %0" :: "n"(N) : "memory");
}

// ---------------------------------------------------------------------------
// WMMA GEMM: C[M,N] = A[M,K] * Bt[N,K]^T, f16 operands, f32 accumulate.
// Block tile 128x128, 4 waves, each wave 64x64 (4x4 WMMA accumulators).
// LDS double-buffered; both tiles K-major (rows padded to 40 halves = 80B,
// so every 16B DMA chunk stays within a row and stays 16B-aligned).
// ---------------------------------------------------------------------------
__device__ __forceinline__ void stage_tile(
    const _Float16* __restrict__ A, const _Float16* __restrict__ Bt,
    int K, int kb, int rowBlock, int colBlock,
    unsigned asBase, unsigned bsBase, int tid) {
#pragma unroll
  for (int u = 0; u < 4; ++u) {
    const int l = tid + u * 128;
    const int row = l >> 2, part = l & 3;
    async_b128(asBase + (unsigned)(row * 80 + part * 16),
               A + (size_t)(rowBlock + row) * K + kb + part * 8);
    async_b128(bsBase + (unsigned)(row * 80 + part * 16),
               Bt + (size_t)(colBlock + row) * K + kb + part * 8);
  }
}

__global__ void __launch_bounds__(128) gemm_wmma_kernel(
    const _Float16* __restrict__ A, const _Float16* __restrict__ Bt,
    float* __restrict__ C, int M, int N, int K) {
  __shared__ _Float16 As[2][128][40];
  __shared__ _Float16 Bs[2][128][40];

  const int tid  = threadIdx.x;
  const int lane = tid & 31;
  const int w    = tid >> 5;   // wave 0..3
  const int wm   = w & 1;      // 2 row tiles of 64
  const int wn   = w >> 1;     // 2 col tiles of 64
  const int rowBlock = blockIdx.y * 128;
  const int colBlock = blockIdx.x * 128;

  const int lr = lane & 15;
  const int kh = (lane >> 4) * 8;   // CDNA5 fragment K-half select

  const unsigned asB[2] = {(unsigned)(uintptr_t)&As[0][0][0],
                           (unsigned)(uintptr_t)&As[1][0][0]};
  const unsigned bsB[2] = {(unsigned)(uintptr_t)&Bs[0][0][0],
                           (unsigned)(uintptr_t)&Bs[1][0][0]};

  v8f acc[4][4];
#pragma unroll
  for (int rt = 0; rt < 4; ++rt)
#pragma unroll
    for (int ct = 0; ct < 4; ++ct) acc[rt][ct] = (v8f){};

  const int nsteps = K / 32;
  stage_tile(A, Bt, K, 0, rowBlock, colBlock, asB[0], bsB[0], tid);

  for (int s = 0; s < nsteps; ++s) {
    const int cur = s & 1;
    if (s + 1 < nsteps) {
      stage_tile(A, Bt, K, (s + 1) * 32, rowBlock, colBlock,
                 asB[cur ^ 1], bsB[cur ^ 1], tid);
      wait_async_le<8>();   // in-order completion: prev tile's 8 copies done
    } else {
      wait_async_le<0>();
    }
    __syncthreads();        // whole block's tile-s staging visible

    v16h a[4], b[4];
#pragma unroll
    for (int t2 = 0; t2 < 4; ++t2) {
      const _Float16* ap = &As[cur][wm * 64 + t2 * 16 + lr][0];
      const _Float16* bp = &Bs[cur][wn * 64 + t2 * 16 + lr][0];
#pragma unroll
      for (int i = 0; i < 8; ++i) {   // contiguous -> ds_load_b128 x2 per frag
        a[t2][i]     = ap[kh + i];
        a[t2][8 + i] = ap[kh + 16 + i];
        b[t2][i]     = bp[kh + i];
        b[t2][8 + i] = bp[kh + 16 + i];
      }
    }
#pragma unroll
    for (int rt = 0; rt < 4; ++rt)
#pragma unroll
      for (int ct = 0; ct < 4; ++ct)
        acc[rt][ct] = __builtin_amdgcn_wmma_f32_16x16x32_f16(
            false, a[rt], false, b[ct], (short)0, acc[rt][ct], false, false);
    __syncthreads();        // all reads of buf[cur] done before it is re-DMA'd
  }

  // C/D layout: VGPR i -> M = i + 8*(lane>=16), N = lane&15
  const int rB = rowBlock + wm * 64 + (lane >> 4) * 8;
  const int cB = colBlock + wn * 64 + lr;
#pragma unroll
  for (int rt = 0; rt < 4; ++rt)
#pragma unroll
    for (int ct = 0; ct < 4; ++ct)
#pragma unroll
      for (int i = 0; i < 8; ++i)
        C[(size_t)(rB + rt * 16 + i) * N + cB + ct * 16] = acc[rt][ct][i];
}

// ---------------------------------------------------------------------------
// f32 -> f16 convert (hidden)
// ---------------------------------------------------------------------------
__global__ void __launch_bounds__(256) cvt_f16_kernel(
    const float* __restrict__ X, _Float16* __restrict__ Y, int n4) {
  int i = (blockIdx.x * 256 + threadIdx.x) * 4;
  if (i < n4 * 4) {
#pragma unroll
    for (int u = 0; u < 4; ++u) Y[i + u] = (_Float16)X[i + u];
  }
}

// ---------------------------------------------------------------------------
// Transpose + convert: Wt[n][k] = (f16) W[k][n]   (W: [R,Ccols] = [K,N])
// ---------------------------------------------------------------------------
__global__ void __launch_bounds__(256) transpose_cvt_kernel(
    const float* __restrict__ W, _Float16* __restrict__ Wt, int R, int Ccols) {
  __shared__ float tile[32][33];
  const int bx = blockIdx.x * 32;   // N
  const int by = blockIdx.y * 32;   // K
  const int tx = threadIdx.x & 31, ty = threadIdx.x >> 5;
#pragma unroll
  for (int r = ty; r < 32; r += 8)
    tile[r][tx] = W[(size_t)(by + r) * Ccols + bx + tx];
  __syncthreads();
#pragma unroll
  for (int r = ty; r < 32; r += 8)
    Wt[(size_t)(bx + r) * R + by + tx] = (_Float16)tile[tx][r];
}

// ---------------------------------------------------------------------------
// beta / g projections (H=16 outputs per token)
// ---------------------------------------------------------------------------
__global__ void __launch_bounds__(256) proj_ab_kernel(
    const float* __restrict__ Xh, const float* __restrict__ Wa,
    const float* __restrict__ Wb, const float* __restrict__ bb,
    const float* __restrict__ A_log, const float* __restrict__ dt_bias,
    float* __restrict__ g, float* __restrict__ beta) {
  const int n   = blockIdx.x;
  const int tid = threadIdx.x;
  const int h   = tid & 15;
  const int chk = tid >> 4;
  const float* x = Xh + (size_t)n * HIDv;
  float sa = 0.f, sb = 0.f;
#pragma unroll 4
  for (int i = 0; i < 64; ++i) {
    float xv = x[chk * 64 + i];
    sa += xv * Wa[(size_t)(chk * 64 + i) * Hh + h];
    sb += xv * Wb[(size_t)(chk * 64 + i) * Hh + h];
  }
  __shared__ float ra[256], rb[256];
  ra[tid] = sa; rb[tid] = sb;
  __syncthreads();
  if (tid < Hh) {
    float ta = 0.f, tb = 0.f;
#pragma unroll
    for (int c = 0; c < 16; ++c) { ta += ra[c * 16 + tid]; tb += rb[c * 16 + tid]; }
    float spx = ta + dt_bias[tid];
    float sp  = (spx > 30.f) ? spx : log1pf(__expf(spx));
    g[(size_t)n * Hh + tid] = -__expf(A_log[tid]) * sp;
    float z = tb + bb[tid];
    beta[(size_t)n * Hh + tid] = 1.f / (1.f + __expf(-z));
  }
}

// ---------------------------------------------------------------------------
// Causal depthwise conv (K=4) + SiLU + per-head (64ch) L2 norm
// ---------------------------------------------------------------------------
__global__ void __launch_bounds__(256) conv_silu_l2_kernel(
    const float* __restrict__ X, const float* __restrict__ Wc,
    float* __restrict__ Out) {
  const int n = blockIdx.x;
  const int b = n / Tv, t = n % Tv;
  const int tid = threadIdx.x;
  __shared__ float sq[256];
  __shared__ float rinv[16];
  float yloc[4];
  float acc2 = 0.f;
#pragma unroll
  for (int u = 0; u < 4; ++u) {
    int c = tid * 4 + u;
    float y = 0.f;
#pragma unroll
    for (int i = 0; i < 4; ++i) {
      int tt = t - 3 + i;
      if (tt >= 0) y += X[((size_t)b * Tv + tt) * HIDv + c] * Wc[c * 4 + i];
    }
    y = y / (1.f + __expf(-y));   // silu
    yloc[u] = y;
    acc2 += y * y;
  }
  sq[tid] = acc2;
  __syncthreads();
  if ((tid & 15) == 0) {
    float s = 0.f;
#pragma unroll
    for (int j = 0; j < 16; ++j) s += sq[tid + j];
    rinv[tid >> 4] = rsqrtf(s + 1e-6f);
  }
  __syncthreads();
  const float rv = rinv[tid >> 4];
#pragma unroll
  for (int u = 0; u < 4; ++u) {
    int c = tid * 4 + u;
    Out[((size_t)b * Tv + t) * HIDv + c] = yloc[u] * rv;
  }
}

// ---------------------------------------------------------------------------
// Phase A: sequential decayed rank-1 prefix; snapshot state at chunk starts.
// snap layout: [bh][chunk][ Hkk(64x64) | Hkv(64x64) ]
// ---------------------------------------------------------------------------
__global__ void __launch_bounds__(64) scan_state_kernel(
    const float* __restrict__ Kn, const float* __restrict__ V,
    const float* __restrict__ g, const float* __restrict__ beta,
    float* __restrict__ snap) {
  const int bh = blockIdx.x;
  const int b = bh >> 4, h = bh & 15;
  const int i = threadIdx.x;
  float Hkk[64];
#pragma unroll
  for (int j = 0; j < 64; ++j) Hkk[j] = 0.f;
  __shared__ float Hkv[64][65];
#pragma unroll
  for (int j = 0; j < 64; ++j) Hkv[i][j] = 0.f;
  __shared__ float kv[64], vv[64];

  for (int c = 0; c < NCHUNK; ++c) {
    float* s = snap + ((size_t)bh * NCHUNK + c) * 2 * 4096;
#pragma unroll
    for (int j = 0; j < 64; ++j) s[i * 64 + j] = Hkk[j];
#pragma unroll
    for (int j = 0; j < 64; ++j) s[4096 + i * 64 + j] = Hkv[i][j];
    for (int tl = 0; tl < CHUNK; ++tl) {
      const int t = c * CHUNK + tl;
      const size_t base = ((size_t)b * Tv + t) * HIDv + h * DK;
      const float ki = Kn[base + i];
      const float vi = V[base + i];
      const float gt = g[((size_t)b * Tv + t) * Hh + h];
      const float bt = beta[((size_t)b * Tv + t) * Hh + h];
      const float decay = __expf(gt);
      kv[i] = ki; vv[i] = vi;
      __syncthreads();
      const float w = bt * ki;
#pragma unroll
      for (int j = 0; j < 64; ++j) Hkk[j] = decay * Hkk[j] + w * kv[j];
#pragma unroll
      for (int j = 0; j < 64; ++j) Hkv[i][j] = decay * Hkv[i][j] + w * vv[j];
      __syncthreads();
    }
  }
}

// 64-thread (2-wave) block sum: wave32 shfl reduce + LDS combine
__device__ __forceinline__ float blk_sum(float v, float* red2, int tid) {
#pragma unroll
  for (int off = 16; off > 0; off >>= 1) v += __shfl_xor(v, off, 32);
  if ((tid & 31) == 0) red2[tid >> 5] = v;
  __syncthreads();
  float s = red2[0] + red2[1];
  __syncthreads();
  return s;
}

// ---------------------------------------------------------------------------
// Phase B: 1024 parallel chunk blocks; per step: rank-1 update, 15-iter CG of
// (Hkk + diag(lamb)) x = q, o = x^T Hkv, RMS norm, emit f16 for final GEMM.
// ---------------------------------------------------------------------------
__global__ void __launch_bounds__(64) scan_cg_kernel(
    const float* __restrict__ Qn, const float* __restrict__ Kn,
    const float* __restrict__ V,  const float* __restrict__ g,
    const float* __restrict__ beta, const float* __restrict__ lambda_params,
    const float* __restrict__ norm_w, const float* __restrict__ snap,
    _Float16* __restrict__ On16) {
  const int blk = blockIdx.x;          // 0..1023
  const int bh = blk / NCHUNK, c = blk % NCHUNK;
  const int b = bh >> 4, h = bh & 15;
  const int i = threadIdx.x;

  const float* s = snap + ((size_t)bh * NCHUNK + c) * 2 * 4096;
  float Hkk[64];
#pragma unroll
  for (int j = 0; j < 64; ++j) Hkk[j] = s[i * 64 + j];
  __shared__ float Hkv[64][65];
#pragma unroll
  for (int j = 0; j < 64; ++j) Hkv[i][j] = s[4096 + i * 64 + j];
  __shared__ float kv[64], vv[64], pv[64], xv[64];
  __shared__ float red2[2];

  const float lraw = lambda_params[h * DK + i];
  const float lamb = ((lraw > 30.f) ? lraw : log1pf(__expf(lraw))) + 0.25f;
  const float nw = norm_w[i];
  __syncthreads();

  for (int tl = 0; tl < CHUNK; ++tl) {
    const int t = c * CHUNK + tl;
    const size_t base = ((size_t)b * Tv + t) * HIDv + h * DK;
    const float qi = Qn[base + i];
    const float ki = Kn[base + i];
    const float vi = V[base + i];
    const float gt = g[((size_t)b * Tv + t) * Hh + h];
    const float bt = beta[((size_t)b * Tv + t) * Hh + h];
    const float decay = __expf(gt);
    kv[i] = ki; vv[i] = vi;
    __syncthreads();
    const float w = bt * ki;
#pragma unroll
    for (int j = 0; j < 64; ++j) Hkk[j] = decay * Hkk[j] + w * kv[j];
#pragma unroll
    for (int j = 0; j < 64; ++j) Hkv[i][j] = decay * Hkv[i][j] + w * vv[j];

    float x = 0.f, r = qi, p = r;
    pv[i] = p;
    float rs = blk_sum(r * r, red2, i);
    for (int it = 0; it < CGIT; ++it) {
      float ap = lamb * p;
#pragma unroll
      for (int j = 0; j < 64; ++j) ap += Hkk[j] * pv[j];   // regs x LDS-broadcast
      const float pAp = blk_sum(p * ap, red2, i);
      const float alpha = rs / (pAp + 1e-12f);
      x += alpha * p;
      r -= alpha * ap;
      const float rs_new = blk_sum(r * r, red2, i);
      p = r + (rs_new / (rs + 1e-12f)) * p;
      pv[i] = p;
      rs = rs_new;
      __syncthreads();
    }
    xv[i] = x;
    __syncthreads();
    float o = 0.f;
#pragma unroll
    for (int j = 0; j < 64; ++j) o += xv[j] * Hkv[j][i];
    const float ss = blk_sum(o * o, red2, i);
    On16[base + i] = (_Float16)(o * rsqrtf(ss * (1.f / 64.f) + 1e-5f) * nw);
    __syncthreads();
  }
}

// ---------------------------------------------------------------------------
extern "C" void kernel_launch(void* const* d_in, const int* in_sizes, int n_in,
                              void* d_out, int out_size, void* d_ws, size_t ws_size,
                              hipStream_t stream) {
  (void)in_sizes; (void)n_in; (void)out_size; (void)ws_size;
  const float* hidden        = (const float*)d_in[0];
  const float* Wq            = (const float*)d_in[1];
  const float* Wk            = (const float*)d_in[2];
  const float* Wv            = (const float*)d_in[3];
  const float* Wa            = (const float*)d_in[4];
  const float* Wb            = (const float*)d_in[5];
  const float* bbv           = (const float*)d_in[6];
  const float* A_log         = (const float*)d_in[7];
  const float* dt_bias       = (const float*)d_in[8];
  const float* lambda_params = (const float*)d_in[9];
  const float* Wconv_q       = (const float*)d_in[10];
  const float* Wconv_k       = (const float*)d_in[11];
  const float* norm_w        = (const float*)d_in[12];
  const float* Wo            = (const float*)d_in[13];
  float* out = (float*)d_out;

  const size_t NTOK = (size_t)Bv * Tv;          // 2048
  const size_t MAT  = NTOK * HIDv;              // 2M elems
  const size_t WMATB = (size_t)HIDv * HIDv * 2; // f16 weight bytes

  char* p = (char*)d_ws;
  auto take = [&](size_t bytes) { char* r = p; p += (bytes + 255) & ~(size_t)255; return r; };
  _Float16* hid16 = (_Float16*)take(MAT * 2);
  _Float16* Wq16  = (_Float16*)take(WMATB);
  _Float16* Wk16  = (_Float16*)take(WMATB);
  _Float16* Wv16  = (_Float16*)take(WMATB);
  _Float16* Wo16  = (_Float16*)take(WMATB);
  float* Xq   = (float*)take(MAT * 4);
  float* Xk   = (float*)take(MAT * 4);
  float* Xv   = (float*)take(MAT * 4);
  float* Qn   = (float*)take(MAT * 4);
  float* Kn   = (float*)take(MAT * 4);
  float* gbuf = (float*)take(NTOK * Hh * 4);
  float* bbuf = (float*)take(NTOK * Hh * 4);
  float* snap = (float*)take((size_t)Bv * Hh * NCHUNK * 2 * 4096 * 4);  // 32 MB
  _Float16* On16 = hid16;   // alias: hid16 dead after the three input GEMMs

  dim3 gblk(128);
  dim3 ggrid(HIDv / 128, NTOK / 128);  // (8,16)
  dim3 tgrid(HIDv / 32, HIDv / 32);    // (32,32)

  cvt_f16_kernel<<<dim3((unsigned)(MAT / 1024)), dim3(256), 0, stream>>>(hidden, hid16, (int)(MAT / 4));
  transpose_cvt_kernel<<<tgrid, dim3(256), 0, stream>>>(Wq, Wq16, HIDv, HIDv);
  transpose_cvt_kernel<<<tgrid, dim3(256), 0, stream>>>(Wk, Wk16, HIDv, HIDv);
  transpose_cvt_kernel<<<tgrid, dim3(256), 0, stream>>>(Wv, Wv16, HIDv, HIDv);
  transpose_cvt_kernel<<<tgrid, dim3(256), 0, stream>>>(Wo, Wo16, HIDv, HIDv);

  gemm_wmma_kernel<<<ggrid, gblk, 0, stream>>>(hid16, Wq16, Xq, (int)NTOK, HIDv, HIDv);
  gemm_wmma_kernel<<<ggrid, gblk, 0, stream>>>(hid16, Wk16, Xk, (int)NTOK, HIDv, HIDv);
  gemm_wmma_kernel<<<ggrid, gblk, 0, stream>>>(hid16, Wv16, Xv, (int)NTOK, HIDv, HIDv);

  proj_ab_kernel<<<dim3((unsigned)NTOK), dim3(256), 0, stream>>>(
      hidden, Wa, Wb, bbv, A_log, dt_bias, gbuf, bbuf);

  conv_silu_l2_kernel<<<dim3((unsigned)NTOK), dim3(256), 0, stream>>>(Xq, Wconv_q, Qn);
  conv_silu_l2_kernel<<<dim3((unsigned)NTOK), dim3(256), 0, stream>>>(Xk, Wconv_k, Kn);

  scan_state_kernel<<<dim3(Bv * Hh), dim3(64), 0, stream>>>(Kn, Xv, gbuf, bbuf, snap);
  scan_cg_kernel<<<dim3(Bv * Hh * NCHUNK), dim3(64), 0, stream>>>(
      Qn, Kn, Xv, gbuf, bbuf, lambda_params, norm_w, snap, On16);

  gemm_wmma_kernel<<<ggrid, gblk, 0, stream>>>(On16, Wo16, out, (int)NTOK, HIDv, HIDv);
}
